// AttentionBlock_75591424409890
// MI455X (gfx1250) — matrix operand order repaired and checked by reading, not compiled, not run on hardware
//
#include <hip/hip_runtime.h>
#include <hip/hip_bf16.h>

// ---------------------------------------------------------------------------
// AttentionBlock on MI455X (gfx1250, wave32, WMMA 16x16x32 f16->f32)
//
//   1) bn_stats:    per-channel mean/rstd over (b,T)
//   2) cvt_w:       qkv_w / proj_w  f32 -> f16
//   3) bn_apply:    xn_t[b][t][c]  (t-major, f16)
//   4) qkv_gemm:    32x32 register-tiled WMMA GEMM -> q_t/k_t [head][t][64]
//                   (pre-scaled), v_c [head][c][1024]
//   5) attn:        per-head attention: hoisted q frags, scores->LDS(f32),
//                   register softmax (wave32 shuffles), probs f16 in-place,
//                   P@V^T with A-frag reuse across 4 accumulators -> a_t
//   6) proj_gemm:   32x32 register-tiled WMMA GEMM + bias + residual -> out
// ---------------------------------------------------------------------------

typedef __attribute__((ext_vector_type(16))) _Float16 v16h;
typedef __attribute__((ext_vector_type(8)))  _Float16 v8h;
typedef __attribute__((ext_vector_type(8)))  float    v8f;

#define WMMA_F32_F16(a, b, c) \
    __builtin_amdgcn_wmma_f32_16x16x32_f16(false, (a), false, (b), (short)0, (c), false, false)

// Combine two contiguous 8-halfword chunks into one 16-halfword fragment.
__device__ __forceinline__ v16h frag_from_chunks(const _Float16* lo_p, const _Float16* hi_p) {
    v8h lo = *(const v8h*)lo_p;
    v8h hi = *(const v8h*)hi_p;
    v16h f;
#pragma unroll
    for (int i = 0; i < 8; ++i) { f[i] = lo[i]; f[i + 8] = hi[i]; }
    return f;
}

// A-matrix 16x32 f16 fragment, row-major source, stride ld (elements).
// Layout: lanes 0-15 hold K=0..15, lanes 16-31 hold K=16..31,
// row M = lane%16.
__device__ __forceinline__ v16h load_frag_a(const _Float16* A, int ld, int lane) {
    const _Float16* p = A + (size_t)(lane & 15) * ld;
    const int kb = (lane < 16) ? 0 : 16;
    return frag_from_chunks(p + kb, p + kb + 8);
}

// B-matrix 32x16 f16 fragment from an N-major (transposed) source Bt[n][k],
// stride ld. Layout (contiguous k, as load_frag_a): lanes 0-15 hold K=0..15, lanes 16-31 hold K=16..31,
// column N = lane%16.
__device__ __forceinline__ v16h load_frag_bt(const _Float16* Bt, int ld, int lane) {
    const _Float16* p = Bt + (size_t)(lane & 15) * ld + ((lane < 16) ? 0 : 16);
    return frag_from_chunks(p, p + 8);
}

// ---------------------------------------------------------------------------
// 1) per-channel mean / rstd over (b=8, T=1024) -> 8192 samples per channel
// ---------------------------------------------------------------------------
__global__ void bn_stats_kernel(const float* __restrict__ x,
                                float* __restrict__ mean, float* __restrict__ rstd) {
    __shared__ float red[2][256];
    const int ch = blockIdx.x;
    float s = 0.f, s2 = 0.f;
    for (int j = threadIdx.x; j < 8192; j += 256) {
        const int b = j >> 10, t = j & 1023;
        const float v = x[((size_t)b * 512 + ch) * 1024 + t];
        s += v; s2 += v * v;
    }
    red[0][threadIdx.x] = s;
    red[1][threadIdx.x] = s2;
    __syncthreads();
    for (int off = 128; off > 0; off >>= 1) {
        if ((int)threadIdx.x < off) {
            red[0][threadIdx.x] += red[0][threadIdx.x + off];
            red[1][threadIdx.x] += red[1][threadIdx.x + off];
        }
        __syncthreads();
    }
    if (threadIdx.x == 0) {
        const float m   = red[0][0] * (1.f / 8192.f);
        const float var = red[1][0] * (1.f / 8192.f) - m * m;
        mean[ch] = m;
        rstd[ch] = rsqrtf(var + 1e-5f);
    }
}

// ---------------------------------------------------------------------------
// 2) weights f32 -> f16
// ---------------------------------------------------------------------------
__global__ void cvt_w_kernel(const float* __restrict__ qw, const float* __restrict__ pw,
                             _Float16* __restrict__ qwh, _Float16* __restrict__ pwh) {
    const int stride = gridDim.x * 256;
    for (int i = blockIdx.x * 256 + threadIdx.x; i < 1536 * 512; i += stride)
        qwh[i] = (_Float16)qw[i];
    for (int i = blockIdx.x * 256 + threadIdx.x; i < 512 * 512; i += stride)
        pwh[i] = (_Float16)pw[i];
}

// ---------------------------------------------------------------------------
// 3) normalize + affine, emit xn_t[b][t][c] in f16 (t-major for GEMM B frags)
// ---------------------------------------------------------------------------
__global__ void bn_apply_kernel(const float* __restrict__ x,
                                const float* __restrict__ mean, const float* __restrict__ rstd,
                                const float* __restrict__ w, const float* __restrict__ bias,
                                _Float16* __restrict__ xn_t) {
    const int bt = blockIdx.x;           // 8192 blocks = b*T
    const int b = bt >> 10, t = bt & 1023;
    for (int c = threadIdx.x; c < 512; c += 256) {
        const float v = x[((size_t)b * 512 + c) * 1024 + t];
        const float n = (v - mean[c]) * rstd[c] * w[c] + bias[c];
        xn_t[((size_t)b * 1024 + t) * 512 + c] = (_Float16)n;
    }
}

// ---------------------------------------------------------------------------
// 4) QKV GEMM: per batch (1536 x 512) @ (512 x 1024)
//    one 32x32 output tile per wave (2x2 WMMA tiles, 4 accumulators):
//    8 b128 loads feed 4 WMMAs per K-step -> 2 VMEM per WMMA.
//    epilogue: +bias, q/k scaled by 64^-0.25, scattered to attention layouts
// ---------------------------------------------------------------------------
#define QK_SCALE 0.35355339059327373f   // 1/sqrt(sqrt(64))

__global__ void qkv_gemm_kernel(const _Float16* __restrict__ W,   // [1536][512]
                                const _Float16* __restrict__ Xt,  // [b][t][512]
                                const float* __restrict__ bias,   // [1536]
                                _Float16* __restrict__ q_t,       // [64 head][1024 t][64 c]
                                _Float16* __restrict__ k_t,       // [64 head][1024 t][64 c]
                                _Float16* __restrict__ v_c) {     // [64 head][64 c][1024 t]
    const int lane = threadIdx.x & 31;
    const int gw = blockIdx.x * 4 + (threadIdx.x >> 5);   // 8*48*32 = 12288 waves
    const int b   = gw / (48 * 32);
    const int rem = gw % (48 * 32);
    const int obase = (rem >> 5) * 32;     // 48 o-tiles of 32
    const int tbase = (rem & 31) * 32;     // 32 t-tiles of 32

    const _Float16* A0 = W  + (size_t)obase * 512;
    const _Float16* B0 = Xt + ((size_t)b * 1024 + tbase) * 512;

    v8f acc[2][2] = {};
#pragma unroll 2
    for (int kk = 0; kk < 512; kk += 32) {
        __builtin_prefetch(A0 + (size_t)(lane & 15) * 512 + kk + 256, 0, 1);
        __builtin_prefetch(B0 + (size_t)(lane & 15) * 512 + kk + 256, 0, 1);
        v16h a0 = load_frag_a (A0 + kk,             512, lane);
        v16h a1 = load_frag_a (A0 + 16 * 512 + kk,  512, lane);
        v16h b0 = load_frag_bt(B0 + kk,             512, lane);
        v16h b1 = load_frag_bt(B0 + 16 * 512 + kk,  512, lane);
        acc[0][0] = WMMA_F32_F16(a0, b0, acc[0][0]);
        acc[0][1] = WMMA_F32_F16(a0, b1, acc[0][1]);
        acc[1][0] = WMMA_F32_F16(a1, b0, acc[1][0]);
        acc[1][1] = WMMA_F32_F16(a1, b1, acc[1][1]);
    }

    const int row_hi = (lane < 16) ? 0 : 8;
#pragma unroll
    for (int mi = 0; mi < 2; ++mi) {
#pragma unroll
        for (int ni = 0; ni < 2; ++ni) {
            const int t = tbase + ni * 16 + (lane & 15);
#pragma unroll
            for (int r = 0; r < 8; ++r) {
                const int o = obase + mi * 16 + r + row_hi;
                const float val = acc[mi][ni][r] + bias[o];
                const int head = o / 192, inner = o % 192;
                const int bh = b * 8 + head;
                if (inner < 64)
                    q_t[((size_t)bh * 1024 + t) * 64 + inner] = (_Float16)(val * QK_SCALE);
                else if (inner < 128)
                    k_t[((size_t)bh * 1024 + t) * 64 + (inner - 64)] = (_Float16)(val * QK_SCALE);
                else
                    v_c[((size_t)bh * 64 + (inner - 128)) * 1024 + t] = (_Float16)val;
            }
        }
    }
}

// ---------------------------------------------------------------------------
// 5) attention per head: each wave owns 16 t-rows.
//    q A-fragments hoisted (loop-invariant over s); scores -> LDS strip;
//    row softmax in registers; probs rewritten in place as f16;
//    P @ V^T with one LDS A-frag shared by 4 accumulators per K-step.
//    2 waves / block -> 128 KB dynamic LDS (of 320 KB per WGP)
// ---------------------------------------------------------------------------
__global__ void attn_kernel(const _Float16* __restrict__ q_t,
                            const _Float16* __restrict__ k_t,
                            const _Float16* __restrict__ v_c,
                            _Float16* __restrict__ a_t) {
    extern __shared__ char smem[];
    const int lane = threadIdx.x & 31;
    const int wave = threadIdx.x >> 5;
    const int bh     = blockIdx.x >> 5;        // 64 heads
    const int rowblk = blockIdx.x & 31;        // 32 blocks of 32 rows
    const int tbase  = rowblk * 32 + wave * 16;
    const int row_hi = (lane < 16) ? 0 : 8;

    float* S = (float*)smem + (size_t)wave * 16 * 1024;   // 16 rows x 1024 f32

    // ---- phase 1: S = (q*s)^T (k*s), K = 64; q frags loop-invariant ----
    const _Float16* qA = q_t + ((size_t)bh * 1024 + tbase) * 64;
    const v16h qa0 = load_frag_a(qA,      64, lane);
    const v16h qa1 = load_frag_a(qA + 32, 64, lane);
    for (int st = 0; st < 64; ++st) {
        const _Float16* kB = k_t + ((size_t)bh * 1024 + st * 16) * 64;
        v16h kb0 = load_frag_bt(kB,      64, lane);
        v16h kb1 = load_frag_bt(kB + 32, 64, lane);
        v8f acc = {};
        acc = WMMA_F32_F16(qa0, kb0, acc);
        acc = WMMA_F32_F16(qa1, kb1, acc);
        const int col = st * 16 + (lane & 15);
#pragma unroll
        for (int r = 0; r < 8; ++r)
            S[(size_t)(r + row_hi) * 1024 + col] = acc[r];
    }

    // ---- phase 2: row softmax; write probs f16 in place ----
    for (int r = 0; r < 16; ++r) {
        float* R = S + (size_t)r * 1024;
        float v[32];
        float m = -3.4e38f;
#pragma unroll
        for (int i = 0; i < 32; ++i) { v[i] = R[lane + 32 * i]; m = fmaxf(m, v[i]); }
        for (int off = 16; off > 0; off >>= 1) m = fmaxf(m, __shfl_xor(m, off, 32));
        float sum = 0.f;
#pragma unroll
        for (int i = 0; i < 32; ++i) { v[i] = __expf(v[i] - m); sum += v[i]; }
        for (int off = 16; off > 0; off >>= 1) sum += __shfl_xor(sum, off, 32);
        const float inv = __builtin_amdgcn_rcpf(sum);
        _Float16* P = (_Float16*)R;   // in place: whole row already in registers
#pragma unroll
        for (int i = 0; i < 32; ++i) P[lane + 32 * i] = (_Float16)(v[i] * inv);
    }

    // ---- phase 3: a^T = P @ v^T, K = 1024; one LDS A-frag -> 4 WMMAs ----
    const _Float16* Ph = (const _Float16*)S;   // [16 t][1024 s] f16, stride 1024
    const int b = bh >> 3;
    const int chead = (bh & 7) * 64;
    const _Float16* vB = v_c + ((size_t)bh * 64 + (lane & 15)) * 1024
                       + ((lane < 16) ? 0 : 16);
    v8f acc[4] = {};
    for (int kk = 0; kk < 1024; kk += 32) {
        v16h a = load_frag_a(Ph + kk, 1024, lane);          // 2x ds_load_b128
#pragma unroll
        for (int ct = 0; ct < 4; ++ct) {
            const _Float16* vp = vB + (size_t)ct * 16 * 1024 + kk;
            v16h bf = frag_from_chunks(vp, vp + 8);
            acc[ct] = WMMA_F32_F16(a, bf, acc[ct]);
        }
    }
#pragma unroll
    for (int ct = 0; ct < 4; ++ct) {
        const int c = chead + ct * 16 + (lane & 15);
#pragma unroll
        for (int r = 0; r < 8; ++r) {
            const int t = tbase + r + row_hi;
            a_t[((size_t)b * 1024 + t) * 512 + c] = (_Float16)acc[ct][r];
        }
    }
}

// ---------------------------------------------------------------------------
// 6) proj GEMM + bias + residual: out = x + proj_w @ a + proj_b   (f32 out)
//    32x32 register tile per wave, 2x2 WMMA tiles.
// ---------------------------------------------------------------------------
__global__ void proj_gemm_kernel(const _Float16* __restrict__ W,   // [512][512]
                                 const _Float16* __restrict__ At,  // [b][t][512]
                                 const float* __restrict__ bias,   // [512]
                                 const float* __restrict__ x,      // [b][512][1024]
                                 float* __restrict__ out) {
    const int lane = threadIdx.x & 31;
    const int gw = blockIdx.x * 4 + (threadIdx.x >> 5);   // 8*16*32 = 4096 waves
    const int b   = gw / (16 * 32);
    const int rem = gw % (16 * 32);
    const int obase = (rem >> 5) * 32;
    const int tbase = (rem & 31) * 32;

    const _Float16* A0 = W  + (size_t)obase * 512;
    const _Float16* B0 = At + ((size_t)b * 1024 + tbase) * 512;

    v8f acc[2][2] = {};
#pragma unroll 2
    for (int kk = 0; kk < 512; kk += 32) {
        __builtin_prefetch(A0 + (size_t)(lane & 15) * 512 + kk + 256, 0, 1);
        __builtin_prefetch(B0 + (size_t)(lane & 15) * 512 + kk + 256, 0, 1);
        v16h a0 = load_frag_a (A0 + kk,            512, lane);
        v16h a1 = load_frag_a (A0 + 16 * 512 + kk, 512, lane);
        v16h b0 = load_frag_bt(B0 + kk,            512, lane);
        v16h b1 = load_frag_bt(B0 + 16 * 512 + kk, 512, lane);
        acc[0][0] = WMMA_F32_F16(a0, b0, acc[0][0]);
        acc[0][1] = WMMA_F32_F16(a0, b1, acc[0][1]);
        acc[1][0] = WMMA_F32_F16(a1, b0, acc[1][0]);
        acc[1][1] = WMMA_F32_F16(a1, b1, acc[1][1]);
    }

    const int row_hi = (lane < 16) ? 0 : 8;
#pragma unroll
    for (int mi = 0; mi < 2; ++mi) {
#pragma unroll
        for (int ni = 0; ni < 2; ++ni) {
            const int t = tbase + ni * 16 + (lane & 15);
#pragma unroll
            for (int r = 0; r < 8; ++r) {
                const int o = obase + mi * 16 + r + row_hi;
                const size_t idx = ((size_t)b * 512 + o) * 1024 + t;
                out[idx] = x[idx] + bias[o] + acc[mi][ni][r];
            }
        }
    }
}

// ---------------------------------------------------------------------------
// launch
// ---------------------------------------------------------------------------
extern "C" void kernel_launch(void* const* d_in, const int* in_sizes, int n_in,
                              void* d_out, int out_size, void* d_ws, size_t ws_size,
                              hipStream_t stream) {
    (void)in_sizes; (void)n_in; (void)out_size; (void)ws_size;

    const float* x      = (const float*)d_in[0];   // (8,512,32,32)
    const float* bn_w   = (const float*)d_in[1];   // (512)
    const float* bn_b   = (const float*)d_in[2];   // (512)
    const float* qkv_w  = (const float*)d_in[3];   // (1536,512)
    const float* qkv_b  = (const float*)d_in[4];   // (1536)
    const float* proj_w = (const float*)d_in[5];   // (512,512)
    const float* proj_b = (const float*)d_in[6];   // (512)
    float* out = (float*)d_out;

    // workspace carve-up (all 256B aligned); total ~44 MB
    char* ws = (char*)d_ws;
    size_t off = 0;
    float* mean = (float*)(ws + off);               // 512 f32
    float* rstd = mean + 512;                       // 512 f32
    off += 4096;
    _Float16* qkvw_h = (_Float16*)(ws + off); off += (size_t)1536 * 512 * 2;
    _Float16* projw_h = (_Float16*)(ws + off); off += (size_t)512 * 512 * 2;
    _Float16* xn_t = (_Float16*)(ws + off); off += (size_t)8 * 1024 * 512 * 2;
    _Float16* q_t  = (_Float16*)(ws + off); off += (size_t)64 * 1024 * 64 * 2;
    _Float16* k_t  = (_Float16*)(ws + off); off += (size_t)64 * 1024 * 64 * 2;
    _Float16* v_c  = (_Float16*)(ws + off); off += (size_t)64 * 64 * 1024 * 2;
    _Float16* a_t  = (_Float16*)(ws + off); off += (size_t)8 * 1024 * 512 * 2;

    bn_stats_kernel<<<512, 256, 0, stream>>>(x, mean, rstd);
    cvt_w_kernel<<<512, 256, 0, stream>>>(qkv_w, proj_w, qkvw_h, projw_h);
    bn_apply_kernel<<<8192, 256, 0, stream>>>(x, mean, rstd, bn_w, bn_b, xn_t);
    qkv_gemm_kernel<<<3072, 128, 0, stream>>>(qkvw_h, xn_t, qkv_b, q_t, k_t, v_c);
    attn_kernel<<<64 * 32, 64, 2 * 16 * 1024 * sizeof(float), stream>>>(q_t, k_t, v_c, a_t);
    proj_gemm_kernel<<<1024, 128, 0, stream>>>(projw_h, a_t, proj_b, x, out);
}
